// SchNetEncoder_8564164789000
// MI455X (gfx1250) — matrix operand order repaired and checked by reading, not compiled, not run on hardware
//
#include <hip/hip_runtime.h>

// SchNet encoder for MI455X (gfx1250, wave32).
// Memory-bound workload -> all GEMM operands f16 (halves edge traffic), f32
// accumulation via v_wmma_f32_16x16x32_f16, weights staged in LDS per block
// (8 waves share one copy; ds_load_b128 path), fused epilogues, f16 edge
// filter, float atomics for the segment_sum.

typedef __attribute__((ext_vector_type(16))) _Float16 v16h;
typedef __attribute__((ext_vector_type(8)))  _Float16 v8h;
typedef __attribute__((ext_vector_type(8)))  float    v8f;

#define LOG2_CONST 0.69314718055994530942f
#define LDS_PAD 8   // halves; rotates bank mapping across columns

__device__ __forceinline__ float ssp_f(float x) {
  // softplus(x) - log(2), stable for large x
  float sp = (x > 20.0f) ? x : __logf(1.0f + __expf(x));
  return sp - LOG2_CONST;
}

__device__ __forceinline__ v16h cat8(v8h lo, v8h hi) {
  return __builtin_shufflevector(lo, hi, 0,1,2,3,4,5,6,7,8,9,10,11,12,13,14,15);
}

// ---------------------------------------------------------------------------
// Block-level WMMA GEMM: C(M x 128) = epilogue(A(M x K) @ B(128 x K)^T + bias)
// A f16 row-major (lda elements); B f16 row-major contiguous (128 x K).
// Block = 256 threads = 8 waves = 64 rows x 128 cols of C. B is staged into
// LDS once per block (padded rows), each wave computes a 16x64 strip with 4
// accumulators sharing one A fragment per 32-wide K step.
// MODE 0: Cf = acc                         (f32, ld=ldc)
// MODE 1: Ch = f16(ssp(acc+b))             (f16, ld=ldc)
// MODE 2: Ch = f16((acc+b)*(elen[row]<=10))(f16, ld=ldc)
// MODE 3: Cf += acc + b                    (residual, ld fixed 128)
// ---------------------------------------------------------------------------
template <int MODE>
__global__ __launch_bounds__(256) void gemm_wmma(
    const _Float16* __restrict__ A, int lda,
    const _Float16* __restrict__ B,
    const float* __restrict__ bias,
    float* __restrict__ Cf,
    _Float16* __restrict__ Ch, int ldc,
    const float* __restrict__ elen,
    int M, int K)
{
  extern __shared__ char smem_raw[];
  _Float16* Bs = (_Float16*)smem_raw;          // 128 x (K + LDS_PAD) halves
  const int ldsw = K + LDS_PAD;

  // --- cooperative stage of B (contiguous 128*K halves) into padded LDS ---
  {
    const int kch = K >> 3;                    // 8-half chunks per row
    const int nchunk = 128 * kch;
    for (int c = threadIdx.x; c < nchunk; c += 256) {
      int row  = c / kch;
      int koff = (c - row * kch) << 3;
      *(v8h*)(Bs + row * ldsw + koff) = *(const v8h*)(B + (c << 3));
    }
  }
  __syncthreads();

  const int wave  = threadIdx.x >> 5;          // 0..7
  const int lane  = threadIdx.x & 31;
  const int strip = wave & 1;                  // cols [0,64) or [64,128)
  const int mtile = blockIdx.x * 4 + (wave >> 1);
  const int m0    = mtile << 4;
  if (m0 >= M) return;
  const int n0    = strip << 6;
  const int r     = lane & 15;
  const int half  = lane >> 4;

  const _Float16* Arow = A + (size_t)(m0 + r) * lda;

  v8f acc0 = {}, acc1 = {}, acc2 = {}, acc3 = {};
  for (int kk = 0; kk < K; kk += 32) {
    // A fragment (16x32 f16): half 0 holds K kk+0..15,
    // half 1 holds K kk+16..31 (2 halves per VGPR).
    v16h a = cat8(*(const v8h*)(Arow + kk + half * 16),
                  *(const v8h*)(Arow + kk + half * 16 + 8));
    // B fragments from LDS: lane = column (n0+t*16+r), K kk+half*16..+15.
    const _Float16* Bk = Bs + kk + half * 16;
    const _Float16* p0 = Bk + (n0 +  0 + r) * ldsw;
    const _Float16* p1 = Bk + (n0 + 16 + r) * ldsw;
    const _Float16* p2 = Bk + (n0 + 32 + r) * ldsw;
    const _Float16* p3 = Bk + (n0 + 48 + r) * ldsw;
    v16h b0 = cat8(*(const v8h*)p0, *(const v8h*)(p0 + 8));
    v16h b1 = cat8(*(const v8h*)p1, *(const v8h*)(p1 + 8));
    v16h b2 = cat8(*(const v8h*)p2, *(const v8h*)(p2 + 8));
    v16h b3 = cat8(*(const v8h*)p3, *(const v8h*)(p3 + 8));
    acc0 = __builtin_amdgcn_wmma_f32_16x16x32_f16(false, a, false, b0, (short)0, acc0, false, false);
    acc1 = __builtin_amdgcn_wmma_f32_16x16x32_f16(false, a, false, b1, (short)0, acc1, false, false);
    acc2 = __builtin_amdgcn_wmma_f32_16x16x32_f16(false, a, false, b2, (short)0, acc2, false, false);
    acc3 = __builtin_amdgcn_wmma_f32_16x16x32_f16(false, a, false, b3, (short)0, acc3, false, false);
  }

  v8f accs[4] = {acc0, acc1, acc2, acc3};
  #pragma unroll
  for (int t = 0; t < 4; ++t) {
    const int col = n0 + t * 16 + r;
    const float bi = bias ? bias[col] : 0.0f;
    #pragma unroll
    for (int v = 0; v < 8; ++v) {
      const int row = m0 + half * 8 + v;       // C/D layout: VGPR v -> row half*8+v
      float val = accs[t][v] + bi;
      if (MODE == 0) {
        Cf[(size_t)row * ldc + col] = val;
      } else if (MODE == 1) {
        Ch[(size_t)row * ldc + col] = (_Float16)ssp_f(val);
      } else if (MODE == 2) {
        float msk = (elen[row] <= 10.0f) ? 1.0f : 0.0f;
        Ch[(size_t)row * ldc + col] = (_Float16)(val * msk);
      } else {  // MODE 3: residual add into h (ld 128), single writer per element
        size_t idx = (size_t)row * 128 + col;
        Cf[idx] = Cf[idx] + val;
      }
    }
  }
}

// h[n,d] = z[n,:5] @ emblin_w[d,:5] + b[d] + z[n,5+d]
__global__ __launch_bounds__(256) void embed_kernel(
    const float* __restrict__ z, const float* __restrict__ w,
    const float* __restrict__ b, float* __restrict__ h, int N)
{
  int gid = blockIdx.x * blockDim.x + threadIdx.x;
  if (gid >= N * 128) return;
  int n = gid >> 7, d = gid & 127;
  const float* zr = z + (size_t)n * 133;
  float acc = b[d];
  #pragma unroll
  for (int k = 0; k < 5; ++k) acc += zr[k] * w[d * 5 + k];
  acc += zr[5 + d];
  h[gid] = acc;
}

// Pad-convert f32 (rows x kin) -> f16 (rows x 128), zeros in [kin,128)
__global__ __launch_bounds__(256) void pad_cvt128(
    const float* __restrict__ in, _Float16* __restrict__ out, int rows, int kin)
{
  int gid = blockIdx.x * blockDim.x + threadIdx.x;
  if (gid >= rows * 128) return;
  int k = gid & 127, row = gid >> 7;
  out[gid] = (k < kin) ? (_Float16)in[(size_t)row * kin + k] : (_Float16)0.0f;
}

__global__ __launch_bounds__(256) void f32_to_f16(
    const float* __restrict__ in, _Float16* __restrict__ out, int n)
{
  int gid = blockIdx.x * blockDim.x + threadIdx.x;
  if (gid < n) out[gid] = (_Float16)in[gid];
}

// h (N x 128 f32) -> hm16[:, 0:128] (row stride 256 f16)
__global__ __launch_bounds__(256) void h_to_hm16(
    const float* __restrict__ h, _Float16* __restrict__ hm, int N)
{
  int gid = blockIdx.x * blockDim.x + threadIdx.x;
  if (gid >= N * 128) return;
  int n = gid >> 7, d = gid & 127;
  hm[(size_t)n * 256 + d] = (_Float16)h[gid];
}

// m_ij = xf[src] * Wf ; m_i = segment_sum(m_ij, dst) via float atomics.
// Wf is f16 (halves the dominant HBM stream); xf gather stays f32 (L2-resident).
__global__ __launch_bounds__(256) void message_scatter(
    const float* __restrict__ xf, const _Float16* __restrict__ Wf,
    const long long* __restrict__ ei, float* __restrict__ mi, int E)
{
  int gid = blockIdx.x * blockDim.x + threadIdx.x;
  if (gid >= E * 128) return;
  int e = gid >> 7, d = gid & 127;
  long long s = ei[e];          // edge_index[0, e]
  long long t = ei[E + e];      // edge_index[1, e]
  float v = xf[s * 128 + d] * (float)Wf[gid];
  atomicAdd(&mi[t * 128 + d], v);
}

extern "C" void kernel_launch(void* const* d_in, const int* in_sizes, int n_in,
                              void* d_out, int out_size, void* d_ws, size_t ws_size,
                              hipStream_t stream) {
  const float*     z     = (const float*)d_in[0];
  const long long* eidx  = (const long long*)d_in[1];
  const float*     elen  = (const float*)d_in[2];
  const float*     eattr = (const float*)d_in[3];
  const float*     emb_w = (const float*)d_in[4];
  const float*     emb_b = (const float*)d_in[5];
  const float*     w1    = (const float*)d_in[6];
  const float*     b1    = (const float*)d_in[7];
  const float*     w2    = (const float*)d_in[8];
  const float*     b2    = (const float*)d_in[9];
  const float*     l1w   = (const float*)d_in[10];
  const float*     l2w   = (const float*)d_in[11];
  const float*     l2b   = (const float*)d_in[12];
  const float*     lw    = (const float*)d_in[13];
  const float*     lb    = (const float*)d_in[14];

  const int E = in_sizes[2];         // 320000
  const int N = in_sizes[0] / 133;   // 20000
  const int L = 6, G = 100;

  float* h = (float*)d_out;          // N x 128, updated in place per layer

  // Workspace carve (~280 MB)
  char* p = (char*)d_ws;
  auto wsalloc = [&](size_t bytes) {
    char* q = p; p += (bytes + 255) & ~(size_t)255; return (void*)q;
  };
  _Float16* ea16 = (_Float16*)wsalloc((size_t)E * 128 * 2);   // padded edge_attr
  _Float16* t116 = (_Float16*)wsalloc((size_t)E * 128 * 2);   // ssp(ea@w1^T+b1)
  _Float16* Wf16 = (_Float16*)wsalloc((size_t)E * 128 * 2);   // edge filter (f16)
  float*    xf   = (float*)   wsalloc((size_t)N * 128 * 4);
  float*    mi   = (float*)   wsalloc((size_t)N * 128 * 4);
  _Float16* mi16 = (_Float16*)wsalloc((size_t)N * 128 * 2);
  _Float16* hm16 = (_Float16*)wsalloc((size_t)N * 256 * 2);   // concat [h, m] f16
  _Float16* w1p  = (_Float16*)wsalloc((size_t)L * 128 * 128 * 2);
  _Float16* w216 = (_Float16*)wsalloc((size_t)L * 128 * 128 * 2);
  _Float16* l116 = (_Float16*)wsalloc((size_t)L * 128 * 128 * 2);
  _Float16* l216 = (_Float16*)wsalloc((size_t)L * 128 * 128 * 2);
  _Float16* lw16 = (_Float16*)wsalloc((size_t)L * 128 * 256 * 2);

  const int TB = 256;
  auto nblk = [](long long n, int tb) { return (int)((n + tb - 1) / tb); };

  // One-time conversions (redone each call; deterministic)
  pad_cvt128<<<nblk((long long)E * 128, TB), TB, 0, stream>>>(eattr, ea16, E, G);
  pad_cvt128<<<nblk((long long)L * 128 * 128, TB), TB, 0, stream>>>(w1, w1p, L * 128, G);
  f32_to_f16<<<nblk((long long)L * 128 * 128, TB), TB, 0, stream>>>(w2, w216, L * 128 * 128);
  f32_to_f16<<<nblk((long long)L * 128 * 128, TB), TB, 0, stream>>>(l1w, l116, L * 128 * 128);
  f32_to_f16<<<nblk((long long)L * 128 * 128, TB), TB, 0, stream>>>(l2w, l216, L * 128 * 128);
  f32_to_f16<<<nblk((long long)L * 128 * 256, TB), TB, 0, stream>>>(lw, lw16, L * 128 * 256);

  embed_kernel<<<nblk((long long)N * 128, TB), TB, 0, stream>>>(z, emb_w, emb_b, h, N);

  // GEMM grids: block = 64 rows (4 mtiles x 2 strips, 8 waves)
  const int gE = (E / 16 + 3) / 4;
  const int gN = (N / 16 + 3) / 4;
  const size_t sh128 = (size_t)128 * (128 + LDS_PAD) * 2;   // 34 KB
  const size_t sh256 = (size_t)128 * (256 + LDS_PAD) * 2;   // 66 KB (320KB LDS/WGP)

  for (int l = 0; l < L; ++l) {
    h_to_hm16<<<nblk((long long)N * 128, TB), TB, 0, stream>>>(h, hm16, N);

    // t1 = ssp(edge_attr @ w1^T + b1)  [f16 out]
    gemm_wmma<1><<<gE, TB, sh128, stream>>>(ea16, 128, w1p + (size_t)l * 128 * 128,
                                            b1 + l * 128, nullptr, t116, 128, nullptr, E, 128);
    // Wf = (t1 @ w2^T + b2) * (edge_length <= 10)  [f16 out]
    gemm_wmma<2><<<gE, TB, sh128, stream>>>(t116, 128, w216 + (size_t)l * 128 * 128,
                                            b2 + l * 128, nullptr, Wf16, 128, elen, E, 128);
    // xf = h @ lin1^T
    gemm_wmma<0><<<gN, TB, sh128, stream>>>(hm16, 256, l116 + (size_t)l * 128 * 128,
                                            nullptr, xf, nullptr, 128, nullptr, N, 128);
    // m_i = segment_sum(xf[src] * Wf, dst)
    hipMemsetAsync(mi, 0, (size_t)N * 128 * 4, stream);
    message_scatter<<<nblk((long long)E * 128, TB), TB, 0, stream>>>(xf, Wf16, eidx, mi, E);
    // m = ssp(m_i @ lin2^T + lin2_b) -> hm16[:, 128:256] (f16, row stride 256)
    f32_to_f16<<<nblk((long long)N * 128, TB), TB, 0, stream>>>(mi, mi16, N * 128);
    gemm_wmma<1><<<gN, TB, sh128, stream>>>(mi16, 128, l216 + (size_t)l * 128 * 128,
                                            l2b + l * 128, nullptr, hm16 + 128, 256, nullptr, N, 128);
    // h += concat([h, m]) @ lin^T + lin_b   (K = 256)
    gemm_wmma<3><<<gN, TB, sh256, stream>>>(hm16, 256, lw16 + (size_t)l * 128 * 256,
                                            lb + l * 128, h, nullptr, 128, nullptr, N, 256);
  }
}
